// CausalSelfAttention_74560632259506
// MI455X (gfx1250) — matrix operand order repaired and checked by reading, not compiled, not run on hardware
//
#include <hip/hip_runtime.h>
#include <stdint.h>

typedef __bf16 bf16_t;
typedef __attribute__((ext_vector_type(16))) __bf16 v16bf;
typedef __attribute__((ext_vector_type(8)))  __bf16 v8bf;
typedef __attribute__((ext_vector_type(8)))  float  v8f;
typedef __attribute__((ext_vector_type(4)))  uint32_t u32x4;
typedef __attribute__((ext_vector_type(8)))  uint32_t u32x8;

#define N_HEADS  16
#define HEAD_DIM 64
#define SEQ_T    2048
#define BATCH    2
#define CDIM     1024

static __device__ __forceinline__ v16bf make_frag(const bf16_t* lo_p, const bf16_t* hi_p) {
  v8bf lo = *(const v8bf*)lo_p;
  v8bf hi = *(const v8bf*)hi_p;
  v16bf r;
#pragma unroll
  for (int i = 0; i < 8; ++i) { r[i] = lo[i]; r[i + 8] = hi[i]; }
  return r;
}

static __device__ __forceinline__ v8f wmma_bf16(v16bf a, v16bf b, v8f c) {
  return __builtin_amdgcn_wmma_f32_16x16x32_bf16(false, a, false, b, (short)0, c, false, false);
}

// async copy 16B global -> LDS (ASYNCcnt-tracked)
static __device__ __forceinline__ void async_b128(uint32_t lds_off, const bf16_t* g) {
  asm volatile("global_load_async_to_lds_b128 %0, %1, off"
               :: "v"(lds_off), "v"((uint64_t)(uintptr_t)g)
               : "memory");
}

// Tensor Data Mover: 2D tile (tile0 x tile1 elements, 2B each) from a
// (dim0 x dim1) row-major tensor into LDS. 2-SGPR-group descriptor (D#).
static __device__ __forceinline__ void tdm_load_2d(uint32_t lds_addr, const void* gaddr,
                                                   uint32_t dim0, uint32_t dim1,
                                                   uint32_t stride0,
                                                   uint32_t tile0, uint32_t tile1) {
  uint64_t ga = (uint64_t)(uintptr_t)gaddr;
  u32x4 g0;
  g0[0] = 1u;                                                  // count = 1 (valid D#)
  g0[1] = lds_addr;                                            // lds_addr [63:32]
  g0[2] = (uint32_t)ga;                                        // global_addr low
  g0[3] = ((uint32_t)(ga >> 32) & 0x01FFFFFFu) | (2u << 30);   // addr hi | type=2
  u32x8 g1;
  g1[0] = 1u << 16;                                            // data_size = 2B
  g1[1] = (dim0 & 0xFFFFu) << 16;                              // tensor_dim0 lo16
  g1[2] = (dim0 >> 16) | ((dim1 & 0xFFFFu) << 16);             // dim0 hi | dim1 lo
  g1[3] = (dim1 >> 16) | (tile0 << 16);                        // dim1 hi | tile_dim0
  g1[4] = tile1;                                               // tile_dim1 (tile_dim2=0)
  g1[5] = stride0;                                             // tensor_dim0_stride lo32
  g1[6] = 0u;
  g1[7] = 0u;
  asm volatile("tensor_load_to_lds %0, %1" :: "s"(g0), "s"(g1) : "memory");
}

// ---------------------------------------------------------------- convert
__global__ void __launch_bounds__(256) cvt_f32_bf16(const float* __restrict__ in,
                                                    bf16_t* __restrict__ out, int n) {
  int i = blockIdx.x * blockDim.x + threadIdx.x;
  if (i < n) out[i] = (bf16_t)in[i];
}

// ---------------------------------------------------------------- GEMM: C[M,N] = A[M,K] * Bt[N,K]^T
__global__ void __launch_bounds__(256) gemm_bf16_nt(const bf16_t* __restrict__ A,
                                                    const bf16_t* __restrict__ Bt,
                                                    float* __restrict__ C,
                                                    int M, int N, int K) {
  __shared__ __align__(16) bf16_t Bs[2][64 * 32];
  const int tid  = threadIdx.x;
  const int lane = tid & 31;
  const int wid  = tid >> 5;
  const int mtiles = M >> 7;
  const int bm = blockIdx.x % mtiles;
  const int bn = blockIdx.x / mtiles;
  const int m0 = bm * 128 + wid * 16;
  const int n0 = bn * 64;

  const int brow = tid >> 2;
  const int bchk = (tid & 3) * 8;
  const bf16_t* gB = Bt + (size_t)(n0 + brow) * K + bchk;
  const uint32_t lB0 = (uint32_t)(uintptr_t)&Bs[0][brow * 32 + bchk];
  const uint32_t lB1 = (uint32_t)(uintptr_t)&Bs[1][brow * 32 + bchk];

  const int m = lane & 15, half = lane >> 4;
  const int kb8  = half ? 8 : 0;
  const int kb16 = half ? 16 : 0;
  const int base8 = half ? 8 : 0;
  const bf16_t* aRow = A + (size_t)(m0 + m) * K;

  async_b128(lB0, gB);
  v16bf a_cur = make_frag(aRow + kb8, aRow + kb8 + 16);

  v8f acc[4] = {};
  const int nsteps = K >> 5;
  for (int s = 0; s < nsteps; ++s) {
    const int cur = s & 1;
    const int knxt = (s + 1 < nsteps) ? (s + 1) * 32 : s * 32;
    if (s + 1 < nsteps) async_b128(cur ? lB0 : lB1, gB + knxt);
    v16bf a_nxt = make_frag(aRow + knxt + kb8, aRow + knxt + kb8 + 16);
    asm volatile("s_wait_asynccnt 0x1" ::: "memory");
    __syncthreads();
    const bf16_t* bbase = &Bs[cur][0];
#pragma unroll
    for (int j = 0; j < 4; ++j) {
      const bf16_t* bp = bbase + (j * 16 + m) * 32 + kb8;
      v16bf b = make_frag(bp, bp + 16);
      acc[j] = wmma_bf16(a_cur, b, acc[j]);
    }
    a_cur = a_nxt;
    __syncthreads();
  }
#pragma unroll
  for (int j = 0; j < 4; ++j)
#pragma unroll
    for (int v = 0; v < 8; ++v)
      C[(size_t)(m0 + v + base8) * N + n0 + j * 16 + m] = acc[j][v];
}

// ---------------------------------------------------------------- RMSNorm + RoPE + V-transpose
__global__ void __launch_bounds__(256) qk_prep(const float* __restrict__ qkv,
                                               const float* __restrict__ qnw,
                                               const float* __restrict__ knw,
                                               bf16_t* __restrict__ qb,
                                               bf16_t* __restrict__ kb,
                                               bf16_t* __restrict__ vt) {
  int tid = blockIdx.x * blockDim.x + threadIdx.x;
  const int total = BATCH * N_HEADS * SEQ_T;
  if (tid >= total) return;
  int t = tid % SEQ_T;
  int h = (tid / SEQ_T) % N_HEADS;
  int b = tid / (SEQ_T * N_HEADS);
  const float* row = qkv + (size_t)(b * SEQ_T + t) * (3 * CDIM);

  float cs[32], sn[32];
#pragma unroll
  for (int i = 0; i < 32; ++i) {
    float inv = __powf(10000.0f, -(float)(2 * i) / 64.0f);
    __sincosf((float)t * inv, &sn[i], &cs[i]);
  }

  auto norm_rope = [&](const float* src, const float* w, bf16_t* dst) {
    float vb[64]; float ss = 0.0f;
#pragma unroll
    for (int d = 0; d < 64; ++d) { vb[d] = src[d]; ss += vb[d] * vb[d]; }
    float rinv = rsqrtf(ss * (1.0f / 64.0f) + 1e-6f);
#pragma unroll
    for (int d = 0; d < 64; ++d) vb[d] = vb[d] * rinv * w[d];
#pragma unroll
    for (int i = 0; i < 32; ++i) {
      float x1 = vb[i], x2 = vb[32 + i];
      dst[i]      = (bf16_t)(x1 * cs[i] - x2 * sn[i]);
      dst[32 + i] = (bf16_t)(x1 * sn[i] + x2 * cs[i]);
    }
  };

  size_t bh = (size_t)(b * N_HEADS + h);
  norm_rope(row + h * HEAD_DIM,        qnw, qb + (bh * SEQ_T + t) * HEAD_DIM);
  norm_rope(row + CDIM + h * HEAD_DIM, knw, kb + (bh * SEQ_T + t) * HEAD_DIM);

  const float* vrow = row + 2 * CDIM + h * HEAD_DIM;
  bf16_t* vbase = vt + bh * HEAD_DIM * SEQ_T;
#pragma unroll
  for (int d = 0; d < 64; ++d) vbase[(size_t)d * SEQ_T + t] = (bf16_t)vrow[d];
}

// ---------------------------------------------------------------- flash attention
// Block = 8 waves = 8 consecutive 16-query tiles of ONE (b,h). K/V tiles are
// DMA'd into LDS by the Tensor Data Mover (wave 0), double-buffered with
// TENSORcnt; all 8 waves consume the shared tiles from LDS.
__global__ void __launch_bounds__(256) attn_fwd(const bf16_t* __restrict__ qb,
                                                const bf16_t* __restrict__ kbuf,
                                                const bf16_t* __restrict__ vt,
                                                bf16_t* __restrict__ ob) {
  __shared__ __align__(16) bf16_t Ks[2][32 * 64];   // [t][d]
  __shared__ __align__(16) bf16_t Vs[2][64 * 32];   // [d][t]
  __shared__ __align__(16) bf16_t plds[8][16 * 32];
  const int lane = threadIdx.x & 31;
  const int wid  = threadIdx.x >> 5;
  const int nblk_per_bh = (SEQ_T / 16) / 8;         // 16
  const int bh   = blockIdx.x / nblk_per_bh;
  const int qblk = blockIdx.x % nblk_per_bh;
  const int q0     = (qblk * 8 + wid) * 16;
  const int q0_max = (qblk * 8 + 7) * 16;
  const int b = bh / N_HEADS;
  const int h = bh % N_HEADS;
  const bf16_t* Q  = qb   + (size_t)bh * SEQ_T * HEAD_DIM;
  const bf16_t* Kp = kbuf + (size_t)bh * SEQ_T * HEAD_DIM;
  const bf16_t* Vp = vt   + (size_t)bh * HEAD_DIM * SEQ_T;

  const int m = lane & 15, half = lane >> 4;
  const int kb8 = half ? 8 : 0, kb16 = half ? 16 : 0, base8 = half ? 8 : 0;

  v16bf qa[2];
#pragma unroll
  for (int kk = 0; kk < 2; ++kk) {
    const bf16_t* p = Q + (size_t)(q0 + m) * HEAD_DIM + kk * 32 + kb8;
    qa[kk] = make_frag(p, p + 16);
  }

  float mrow[8], lrow[8];
#pragma unroll
  for (int v = 0; v < 8; ++v) { mrow[v] = -3.0e38f; lrow[v] = 0.0f; }
  v8f o[4] = {};

  const uint32_t ldsK[2] = { (uint32_t)(uintptr_t)&Ks[0][0], (uint32_t)(uintptr_t)&Ks[1][0] };
  const uint32_t ldsV[2] = { (uint32_t)(uintptr_t)&Vs[0][0], (uint32_t)(uintptr_t)&Vs[1][0] };

  if (wid == 0) {   // prologue: tiles for s0 = 0 into buffer 0
    tdm_load_2d(ldsK[0], Kp, HEAD_DIM, SEQ_T, HEAD_DIM, HEAD_DIM, 32); // 32 K-rows x 64d
    tdm_load_2d(ldsV[0], Vp, SEQ_T, HEAD_DIM, SEQ_T, 32, HEAD_DIM);    // 64 d-rows x 32t
  }

  for (int s0 = 0; s0 <= q0_max; s0 += 32) {
    const int cur = (s0 >> 5) & 1, nxt = cur ^ 1;
    const int sn = (s0 + 32 <= q0_max) ? s0 + 32 : s0;
    if (wid == 0) {  // prefetch next tiles via TDM
      tdm_load_2d(ldsK[nxt], Kp + (size_t)sn * HEAD_DIM, HEAD_DIM, SEQ_T, HEAD_DIM, HEAD_DIM, 32);
      tdm_load_2d(ldsV[nxt], Vp + sn, SEQ_T, HEAD_DIM, SEQ_T, 32, HEAD_DIM);
    }
    __builtin_amdgcn_s_wait_tensorcnt(2);  // in-order: current pair has landed
    __syncthreads();

    if (s0 <= q0) {  // wave-uniform causal skip
      // S = Q * K^T (16 x 32) from LDS K tile
      v8f sf[2] = {};
#pragma unroll
      for (int j = 0; j < 2; ++j)
#pragma unroll
        for (int kk = 0; kk < 2; ++kk) {
          const bf16_t* p = &Ks[cur][(j * 16 + m) * 64 + kk * 32 + kb8];
          sf[j] = wmma_bf16(qa[kk], make_frag(p, p + 16), sf[j]);
        }

      const bool diag = (s0 + 32 > q0);
#pragma unroll
      for (int j = 0; j < 2; ++j)
#pragma unroll
        for (int v = 0; v < 8; ++v) {
          float val = sf[j][v] * 0.125f;
          if (diag && (s0 + j * 16 + m) > (q0 + v + base8)) val = -1.0e30f;
          sf[j][v] = val;
        }

      float corr[8];
#pragma unroll
      for (int v = 0; v < 8; ++v) {
        float vm = fmaxf(sf[0][v], sf[1][v]);
        vm = fmaxf(vm, __shfl_xor(vm, 1, 32));
        vm = fmaxf(vm, __shfl_xor(vm, 2, 32));
        vm = fmaxf(vm, __shfl_xor(vm, 4, 32));
        vm = fmaxf(vm, __shfl_xor(vm, 8, 32));
        float mnew = fmaxf(mrow[v], vm);
        corr[v] = __expf(mrow[v] - mnew);
        mrow[v] = mnew;
        float p0 = __expf(sf[0][v] - mnew);
        float p1 = __expf(sf[1][v] - mnew);
        sf[0][v] = p0; sf[1][v] = p1;
        float rs = p0 + p1;
        rs += __shfl_xor(rs, 1, 32);
        rs += __shfl_xor(rs, 2, 32);
        rs += __shfl_xor(rs, 4, 32);
        rs += __shfl_xor(rs, 8, 32);
        lrow[v] = lrow[v] * corr[v] + rs;
      }
#pragma unroll
      for (int d = 0; d < 4; ++d)
#pragma unroll
        for (int v = 0; v < 8; ++v) o[d][v] *= corr[v];

      // D-fragment -> A-fragment transpose through wave-private LDS
      bf16_t* pl = plds[wid];
#pragma unroll
      for (int j = 0; j < 2; ++j)
#pragma unroll
        for (int v = 0; v < 8; ++v)
          pl[(v + base8) * 32 + j * 16 + m] = (bf16_t)sf[j][v];
      asm volatile("s_wait_dscnt 0x0" ::: "memory");
      v16bf pa = make_frag(pl + m * 32 + kb8, pl + m * 32 + 16 + kb8);

      // O += P(16x32) * V(32x64) from LDS V tile
#pragma unroll
      for (int d = 0; d < 4; ++d) {
        const bf16_t* p = &Vs[cur][(d * 16 + m) * 32 + kb8];
        o[d] = wmma_bf16(pa, make_frag(p, p + 16), o[d]);
      }
    }
    __syncthreads();   // protect buffer reuse
  }

#pragma unroll
  for (int v = 0; v < 8; ++v) lrow[v] = 1.0f / lrow[v];
#pragma unroll
  for (int d = 0; d < 4; ++d)
#pragma unroll
    for (int v = 0; v < 8; ++v) {
      int t = q0 + v + base8;
      ob[(size_t)(b * SEQ_T + t) * (N_HEADS * HEAD_DIM) + h * HEAD_DIM + d * 16 + m] =
          (bf16_t)(o[d][v] * lrow[v]);
    }
}

// ---------------------------------------------------------------- host
extern "C" void kernel_launch(void* const* d_in, const int* in_sizes, int n_in,
                              void* d_out, int out_size, void* d_ws, size_t ws_size,
                              hipStream_t stream) {
  const float* x    = (const float*)d_in[0];
  const float* wqkv = (const float*)d_in[1];
  const float* wo   = (const float*)d_in[2];
  const float* qnw  = (const float*)d_in[3];
  const float* knw  = (const float*)d_in[4];
  float* out = (float*)d_out;

  char* ws = (char*)d_ws;
  size_t off = 0;
  auto take = [&](size_t bytes) -> char* {
    char* p = ws + off;
    off = (off + bytes + 255) & ~(size_t)255;
    return p;
  };

  const size_t M = (size_t)BATCH * SEQ_T;                      // 4096
  bf16_t* x_b    = (bf16_t*)take(M * CDIM * 2);
  bf16_t* wqkv_b = (bf16_t*)take((size_t)3 * CDIM * CDIM * 2);
  bf16_t* wo_b   = (bf16_t*)take((size_t)CDIM * CDIM * 2);
  float*  qkv_f  = (float*)take(M * 3 * CDIM * 4);
  bf16_t* q_b    = (bf16_t*)take(M * CDIM * 2);
  bf16_t* k_b    = (bf16_t*)take(M * CDIM * 2);
  bf16_t* vt_b   = (bf16_t*)take(M * CDIM * 2);
  bf16_t* attn_b = (bf16_t*)qkv_f;                             // alias dead qkv buffer

  int nx = (int)(M * CDIM);
  cvt_f32_bf16<<<(nx + 255) / 256, 256, 0, stream>>>(x, x_b, nx);
  int nw = 3 * CDIM * CDIM;
  cvt_f32_bf16<<<(nw + 255) / 256, 256, 0, stream>>>(wqkv, wqkv_b, nw);
  int no = CDIM * CDIM;
  cvt_f32_bf16<<<(no + 255) / 256, 256, 0, stream>>>(wo, wo_b, no);

  gemm_bf16_nt<<<(4096 / 128) * (3072 / 64), 256, 0, stream>>>(x_b, wqkv_b, qkv_f, 4096, 3 * CDIM, CDIM);

  qk_prep<<<(BATCH * N_HEADS * SEQ_T) / 256, 256, 0, stream>>>(qkv_f, qnw, knw, q_b, k_b, vt_b);

  // 512 blocks: 32 (b,h) x 16 q-blocks of 128 queries
  attn_fwd<<<BATCH * N_HEADS * ((SEQ_T / 16) / 8), 256, 0, stream>>>(q_b, k_b, vt_b, attn_b);

  gemm_bf16_nt<<<(4096 / 128) * (1024 / 64), 256, 0, stream>>>(attn_b, wo_b, out, 4096, CDIM, CDIM);
}
